// MultiHeadAttention_66984309949014
// MI455X (gfx1250) — matrix operand order repaired and checked by reading, not compiled, not run on hardware
//
#include <hip/hip_runtime.h>
#include <hip/hip_bf16.h>

// MHA: B=2, T=2048, E=1024, H=16, HD=64.  M = B*T = 4096.
// d_ws layout (f16): Q[4096][1024] | K[4096][1024] | V[4096][1024]  (24 MB)

typedef _Float16 v16h __attribute__((ext_vector_type(16)));
typedef _Float16 v8h  __attribute__((ext_vector_type(8)));
typedef float    v8f  __attribute__((ext_vector_type(8)));

#define MT 4096
#define ET 1024
#define TT 2048

#if defined(__has_builtin)
#if __has_builtin(__builtin_amdgcn_global_load_async_to_lds_b128) && \
    __has_builtin(__builtin_amdgcn_s_wait_asynccnt)
#define USE_ASYNC_COPY 1
#endif
#endif

#ifdef USE_ASYNC_COPY
typedef int v4i __attribute__((vector_size(4 * sizeof(int))));
typedef __attribute__((address_space(1))) v4i* gv4i_ptr;   // global src
typedef __attribute__((address_space(3))) v4i* lv4i_ptr;   // LDS dst
#endif

static __device__ __forceinline__ float fast_exp2(float x) {
#if __has_builtin(__builtin_amdgcn_exp2f)
  return __builtin_amdgcn_exp2f(x);
#else
  return exp2f(x);
#endif
}
static __device__ __forceinline__ float fast_rcp(float x) {
#if __has_builtin(__builtin_amdgcn_rcpf)
  return __builtin_amdgcn_rcpf(x);
#else
  return 1.0f / x;
#endif
}
static __device__ __forceinline__ v16h cat16(v8h lo, v8h hi) {
  return __builtin_shufflevector(lo, hi, 0,1,2,3,4,5,6,7,8,9,10,11,12,13,14,15);
}
static __device__ __forceinline__ v8f wmma_f16(v16h a, v16h b, v8f c) {
  return __builtin_amdgcn_wmma_f32_16x16x32_f16(false, a, false, b, (short)0, c,
                                                false, false);
}

// ---------------------------------------------------------------------------
// Kernel 1: C = x @ W  (fp32 in, f16 out into workspace). grid.z picks Q/K/V.
// Block: 256 thr (8 waves). Block tile 128(M) x 128(N); wave tile 32 x 64.
// ---------------------------------------------------------------------------
__global__ __launch_bounds__(256) void qkv_gemm_kernel(
    const float* __restrict__ x, const float* __restrict__ wq,
    const float* __restrict__ wk, const float* __restrict__ wv,
    _Float16* __restrict__ ws) {
  __shared__ _Float16 xs[128 * 32];   // x tile, row-major [m][k], stride 32
  __shared__ _Float16 wt[128 * 32];   // W tile transposed [n][k], stride 32

  const int which = blockIdx.z;
  const float* W = (which == 0) ? wq : (which == 1) ? wk : wv;
  _Float16* out = ws + (size_t)which * ((size_t)MT * ET);

  const int n0 = blockIdx.x * 128;
  const int m0 = blockIdx.y * 128;
  const int tid  = threadIdx.x;
  const int lane = tid & 31;
  const int wid  = tid >> 5;
  const int wave_m = (wid & 3) * 32;
  const int wave_n = (wid >> 2) * 64;
  const int lm = lane & 15;   // row (A) / col (B,C) within 16
  const int hh = lane >> 4;   // half-wave select

  v8f c[2][4];
#pragma unroll
  for (int i = 0; i < 2; ++i)
#pragma unroll
    for (int j = 0; j < 4; ++j) c[i][j] = (v8f){0.f,0.f,0.f,0.f,0.f,0.f,0.f,0.f};

  for (int k0 = 0; k0 < ET; k0 += 32) {
    __syncthreads();
    // stage x: 128 rows x 32 k (coalesced global, f32 -> f16)
    {
      const int col = tid & 31;
      const int rb  = tid >> 5;
#pragma unroll
      for (int i = 0; i < 16; ++i) {
        const int r = rb + i * 8;
        xs[r * 32 + col] = (_Float16)x[(size_t)(m0 + r) * ET + k0 + col];
      }
    }
    // stage W transposed: wt[n][k], 32(k) x 128(n)
    {
      const int n  = tid & 127;
      const int kb = tid >> 7;
#pragma unroll
      for (int i = 0; i < 16; ++i) {
        const int k = kb + i * 2;
        wt[n * 32 + k] = (_Float16)W[(size_t)(k0 + k) * ET + n0 + n];
      }
    }
    __syncthreads();

    v16h a[2], bfr[4];
#pragma unroll
    for (int mm = 0; mm < 2; ++mm) {
      // A 16x32 layout: elem i <-> K = 16*(i/8) + 8*hh + (i%8)
      const _Float16* base = &xs[(wave_m + mm * 16 + lm) * 32 + 8 * hh];
      a[mm] = cat16(*(const v8h*)(base), *(const v8h*)(base + 16));
    }
#pragma unroll
    for (int nn = 0; nn < 4; ++nn) {
      // B 32x16 layout: elem i <-> K = 16*(i/8) + 8*hh + (i%8)  (wt is [n][k])
      const _Float16* base = &wt[(wave_n + nn * 16 + lm) * 32 + 8 * hh];
      bfr[nn] = cat16(*(const v8h*)(base), *(const v8h*)(base + 16));
    }
#pragma unroll
    for (int mm = 0; mm < 2; ++mm)
#pragma unroll
      for (int nn = 0; nn < 4; ++nn)
        c[mm][nn] = wmma_f16(a[mm], bfr[nn], c[mm][nn]);
  }

  // store f16 result (C layout: VGPR r -> row r + 8*hh, lane -> col lm)
#pragma unroll
  for (int mm = 0; mm < 2; ++mm)
#pragma unroll
    for (int nn = 0; nn < 4; ++nn)
#pragma unroll
      for (int r = 0; r < 8; ++r) {
        const int gm = m0 + wave_m + mm * 16 + r + 8 * hh;
        const int gn = n0 + wave_n + nn * 16 + lm;
        out[(size_t)gm * ET + gn] = (_Float16)c[mm][nn][r];
      }
}

// ---------------------------------------------------------------------------
// Kernel 2: flash attention per (b,h). Block: 8 waves x 16 queries = 128 q.
// K staged row-major [key][d] (async-to-LDS when available); V staged
// transposed [d][key]; P re-laid out through per-wave LDS buffer
// (C-frag -> A-frag); row sums via WMMA against an all-ones B matrix.
// ---------------------------------------------------------------------------
__global__ __launch_bounds__(256) void flash_attn_kernel(
    const _Float16* __restrict__ ws, float* __restrict__ out) {
  __shared__ _Float16 ks[128 * 64];   // [key][d], stride 64
  __shared__ _Float16 vt[64 * 128];   // [d][key], stride 128
  __shared__ _Float16 ps[8][16 * 32]; // per-wave P buffer [row][col], stride 32

  const _Float16* qw = ws;
  const _Float16* kw = ws + (size_t)MT * ET;
  const _Float16* vw = ws + (size_t)2 * MT * ET;

  const int bid  = blockIdx.x;        // [qblk(16)][h(16)][b(2)]
  const int qblk = bid & 15;
  const int h    = (bid >> 4) & 15;
  const int b    = bid >> 8;
  const int tid  = threadIdx.x;
  const int lane = tid & 31;
  const int wid  = tid >> 5;
  const int lm   = lane & 15;
  const int hh   = lane >> 4;

  const int qr0 = qblk * 128 + wid * 16;        // query row base (in T)

  // Load Q A-fragments, pre-scaled by (1/sqrt(64)) * log2(e)
  const _Float16 qscale = (_Float16)(0.125f * 1.44269504088896f);
  v16h qa[2];
#pragma unroll
  for (int cc = 0; cc < 2; ++cc) {
    const _Float16* base = qw + (size_t)(b * TT + qr0 + lm) * ET + h * 64 +
                           32 * cc + 8 * hh;
    qa[cc] = cat16(*(const v8h*)(base), *(const v8h*)(base + 16));
    qa[cc] *= qscale;
  }

  const v8f zero8 = (v8f){0.f,0.f,0.f,0.f,0.f,0.f,0.f,0.f};
  v16h ones;
#pragma unroll
  for (int i = 0; i < 16; ++i) ones[i] = (_Float16)1.0f;

  v8f o[4];
#pragma unroll
  for (int f = 0; f < 4; ++f) o[f] = zero8;
  float mrow[8], lrow[8];
#pragma unroll
  for (int r = 0; r < 8; ++r) { mrow[r] = -1.0e30f; lrow[r] = 0.0f; }

  for (int kb = 0; kb < TT; kb += 128) {
    __syncthreads();
    // stage K (row-major, async when available) and V (transposed)
    {
      const int d0  = (tid & 7) * 8;
      const int krb = tid >> 3;            // 0..31
#pragma unroll
      for (int i = 0; i < 4; ++i) {
        const int kr = krb + 32 * i;
        const size_t g = (size_t)(b * TT + kb + kr) * ET + h * 64 + d0;
#ifdef USE_ASYNC_COPY
        __builtin_amdgcn_global_load_async_to_lds_b128(
            (gv4i_ptr)(kw + g), (lv4i_ptr)&ks[kr * 64 + d0], 0, 0);
#else
        *(v8h*)&ks[kr * 64 + d0] = *(const v8h*)(kw + g);
#endif
        const v8h vv = *(const v8h*)(vw + g);
#pragma unroll
        for (int j = 0; j < 8; ++j) vt[(d0 + j) * 128 + kr] = vv[j];
      }
      if (kb + 128 < TT) {  // hint next K block toward L2
        __builtin_prefetch(kw + (size_t)(b * TT + kb + 128 + krb) * ET + h * 64, 0, 1);
      }
    }
#ifdef USE_ASYNC_COPY
    __builtin_amdgcn_s_wait_asynccnt(0);
#endif
    __syncthreads();

#pragma unroll
    for (int jt = 0; jt < 4; ++jt) {      // 32 keys per inner step
      const int kk0 = jt * 32;

      // S = Q @ K^T : two 16x16 C-frags (keys kk0..+15, kk0+16..+31)
      v8f s[2];
#pragma unroll
      for (int sI = 0; sI < 2; ++sI) {
        s[sI] = zero8;
#pragma unroll
        for (int cc = 0; cc < 2; ++cc) {
          const _Float16* base =
              &ks[(kk0 + sI * 16 + lm) * 64 + 32 * cc + 8 * hh];
          const v16h bf = cat16(*(const v8h*)(base), *(const v8h*)(base + 16));
          s[sI] = wmma_f16(qa[cc], bf, s[sI]);
        }
      }

      // online softmax: row max via cross-lane reduce, P scattered to LDS
      float corr[8];
#pragma unroll
      for (int r = 0; r < 8; ++r) {
        float t = fmaxf(s[0][r], s[1][r]);
        t = fmaxf(t, __shfl_xor(t, 1, 32));
        t = fmaxf(t, __shfl_xor(t, 2, 32));
        t = fmaxf(t, __shfl_xor(t, 4, 32));
        t = fmaxf(t, __shfl_xor(t, 8, 32));
        const float mn = fmaxf(mrow[r], t);
        corr[r] = fast_exp2(mrow[r] - mn);
        mrow[r] = mn;
        const float e0 = fast_exp2(s[0][r] - mn);
        const float e1 = fast_exp2(s[1][r] - mn);
        const int prow = r + 8 * hh;
        ps[wid][prow * 32 + lm]      = (_Float16)e0;
        ps[wid][prow * 32 + 16 + lm] = (_Float16)e1;
#pragma unroll
        for (int f = 0; f < 4; ++f) o[f][r] *= corr[r];
      }

      // re-load P as A-fragment (16x32)
      const _Float16* pb = &ps[wid][lm * 32 + 8 * hh];
      const v16h pa = cat16(*(const v8h*)(pb), *(const v8h*)(pb + 16));

      // row sums of P via WMMA against all-ones B (replicated per C layout)
      const v8f rs = wmma_f16(pa, ones, zero8);
#pragma unroll
      for (int r = 0; r < 8; ++r) lrow[r] = lrow[r] * corr[r] + rs[r];

      // O += P @ V : 4 d-tiles of 16
#pragma unroll
      for (int nt = 0; nt < 4; ++nt) {
        const _Float16* vb = &vt[(nt * 16 + lm) * 128 + kk0 + 8 * hh];
        const v16h bf = cat16(*(const v8h*)(vb), *(const v8h*)(vb + 16));
        o[nt] = wmma_f16(pa, bf, o[nt]);
      }
    }
  }

  // normalize and store fp32 output [B,T,E]
#pragma unroll
  for (int r = 0; r < 8; ++r) {
    const float inv = fast_rcp(lrow[r]);
    const int gq = qr0 + r + 8 * hh;
#pragma unroll
    for (int nt = 0; nt < 4; ++nt) {
      out[(size_t)(b * TT + gq) * ET + h * 64 + nt * 16 + lm] = o[nt][r] * inv;
    }
  }
}

// ---------------------------------------------------------------------------
extern "C" void kernel_launch(void* const* d_in, const int* in_sizes, int n_in,
                              void* d_out, int out_size, void* d_ws,
                              size_t ws_size, hipStream_t stream) {
  (void)in_sizes; (void)n_in; (void)out_size; (void)ws_size;
  const float* x  = (const float*)d_in[0];
  const float* wq = (const float*)d_in[1];
  const float* wk = (const float*)d_in[2];
  const float* wv = (const float*)d_in[3];
  _Float16* wsp = (_Float16*)d_ws;
  float* outp = (float*)d_out;

  // QKV projections: grid (N/128, M/128, 3)
  dim3 g1(ET / 128, MT / 128, 3);
  qkv_gemm_kernel<<<g1, 256, 0, stream>>>(x, wq, wk, wv, wsp);

  // Flash attention: 512 blocks = (T/128) * H * B
  flash_attn_kernel<<<dim3(512, 1, 1), 256, 0, stream>>>(wsp, outp);
}
